// LegacyDense_4698694222453
// MI455X (gfx1250) — matrix operand order repaired and checked by reading, not compiled, not run on hardware
//
#include <hip/hip_runtime.h>

// ---------------------------------------------------------------------------
// SNN forward: h1 = x @ w  (M=32000, N=512, K=700, f32 in/out) then LIF scan.
//
// GEMM: v_wmma_f32_16x16x32_bf16 (bf16 in, f32 accum), double-buffered LDS,
// global loads for chunk k+1 in flight under the WMMAs of chunk k.
// Scan: pure streaming, ~131 MB @ 23.3 TB/s.
// ---------------------------------------------------------------------------

typedef __attribute__((ext_vector_type(16))) __bf16 v16bf;
typedef __attribute__((ext_vector_type(8)))  float  v8f;

namespace {
constexpr int Bsz = 128;
constexpr int T   = 250;
constexpr int I   = 700;
constexpr int O   = 512;
constexpr int M   = Bsz * T;        // 32000 rows of h1

constexpr int TILE_M = 128;
constexpr int TILE_N = 128;
constexpr int TILE_K = 32;
constexpr int KCHUNKS = (I + TILE_K - 1) / TILE_K;   // 22 (last has 28 valid)

constexpr float ALPHA = 0.8187307530779818f;  // exp(-dt/tau_syn)
constexpr float BETA  = 0.9048374180359595f;  // exp(-dt/tau_mem)
}

// Pack two f32 into a dword of bf16 (round-to-nearest-even).
__device__ __forceinline__ unsigned int pk_bf16(float a, float b) {
#if __has_builtin(__builtin_amdgcn_cvt_pk_bf16_f32)
    typedef __attribute__((ext_vector_type(2))) __bf16 v2bf;
    v2bf r = __builtin_amdgcn_cvt_pk_bf16_f32(a, b);
    return __builtin_bit_cast(unsigned int, r);
#else
    unsigned short ua = __builtin_bit_cast(unsigned short, (__bf16)a);
    unsigned short ub = __builtin_bit_cast(unsigned short, (__bf16)b);
    return (unsigned int)ua | ((unsigned int)ub << 16);
#endif
}

// ---------------------------------------------------------------------------
// GEMM: 256 threads = 8 wave32, block tile 128x128, waves 4(M) x 2(N),
// each wave 32x64 of C -> 2x4 accumulators, 8 WMMAs per 32-wide K chunk.
// ---------------------------------------------------------------------------
__global__ __launch_bounds__(256) void snn_gemm_bf16(
        const float* __restrict__ x,
        const float* __restrict__ w,
        float* __restrict__ h1) {
    // Double-buffered: A [row][k] bf16 (8KB each), B transposed [col][k]
    // bf16 (8KB each) so fragment loads are contiguous-K ds_load_b128.
    __shared__ __align__(16) unsigned short lds_a[2][TILE_M * TILE_K];
    __shared__ __align__(16) unsigned short lds_b[2][TILE_N * TILE_K];

    const int tid  = threadIdx.x;
    const int m0   = blockIdx.y * TILE_M;
    const int n0   = blockIdx.x * TILE_N;
    const int wave = tid >> 5;
    const int lane = tid & 31;
    const int l16  = lane & 15;
    const int lh   = lane >> 4;
    const int wm   = (wave >> 1) * 32;   // wave M offset (0,32,64,96)
    const int wn   = (wave & 1) * 64;    // wave N offset (0,64)

    // Staging registers for the in-flight chunk.
    float4 ra[4];                         // A: 4 float4 per thread
    float  rb[16];                        // B: one column, 16 K values
    const int bcol = tid & 127;           // B staging: this thread's column
    const int bkb  = (tid >> 7) * 8;      // and K sub-base (0 or 8)

    auto load_chunk = [&](int k0) {
        if (k0 + TILE_K <= I) {
#pragma unroll
            for (int i = 0; i < 4; ++i) {
                const int slot = tid + i * 256;          // 0..1023
                const int row  = slot >> 3;              // 0..127
                const int kq   = (slot & 7) << 2;        // 0..28
                ra[i] = *(const float4*)(x + (size_t)(m0 + row) * I + k0 + kq);
            }
#pragma unroll
            for (int p = 0; p < 2; ++p)
#pragma unroll
                for (int j = 0; j < 8; ++j)
                    rb[p * 8 + j] = w[(size_t)(k0 + bkb + p * 16 + j) * O + n0 + bcol];
        } else {                                          // K tail: zero-fill
#pragma unroll
            for (int i = 0; i < 4; ++i) {
                const int slot = tid + i * 256;
                const int row  = slot >> 3;
                const int kq   = (slot & 7) << 2;
                const float* src = x + (size_t)(m0 + row) * I + k0 + kq;
                ra[i].x = (k0 + kq + 0 < I) ? src[0] : 0.f;
                ra[i].y = (k0 + kq + 1 < I) ? src[1] : 0.f;
                ra[i].z = (k0 + kq + 2 < I) ? src[2] : 0.f;
                ra[i].w = (k0 + kq + 3 < I) ? src[3] : 0.f;
            }
#pragma unroll
            for (int p = 0; p < 2; ++p)
#pragma unroll
                for (int j = 0; j < 8; ++j) {
                    const int k = k0 + bkb + p * 16 + j;
                    rb[p * 8 + j] = (k < I) ? w[(size_t)k * O + n0 + bcol] : 0.f;
                }
        }
    };

    auto store_chunk = [&](int buf) {
#pragma unroll
        for (int i = 0; i < 4; ++i) {                    // A: ds_store_b64
            const int slot = tid + i * 256;
            const int row  = slot >> 3;
            const int kq   = (slot & 7) << 2;
            uint2 p;
            p.x = pk_bf16(ra[i].x, ra[i].y);
            p.y = pk_bf16(ra[i].z, ra[i].w);
            *(uint2*)(&lds_a[buf][row * TILE_K + kq]) = p;
        }
#pragma unroll
        for (int p = 0; p < 2; ++p) {                    // B: ds_store_b128
            uint4 q;
            q.x = pk_bf16(rb[p * 8 + 0], rb[p * 8 + 1]);
            q.y = pk_bf16(rb[p * 8 + 2], rb[p * 8 + 3]);
            q.z = pk_bf16(rb[p * 8 + 4], rb[p * 8 + 5]);
            q.w = pk_bf16(rb[p * 8 + 6], rb[p * 8 + 7]);
            *(uint4*)(&lds_b[buf][bcol * TILE_K + bkb + p * 16]) = q;
        }
    };

    v8f acc[2][4];
#pragma unroll
    for (int mi = 0; mi < 2; ++mi)
#pragma unroll
        for (int ni = 0; ni < 4; ++ni) acc[mi][ni] = (v8f){};

    // Prologue: stage chunk 0.
    load_chunk(0);
    store_chunk(0);
    __syncthreads();

    for (int kc = 0; kc < KCHUNKS; ++kc) {
        const int cur = kc & 1;

        // Issue next chunk's global loads before the WMMAs (latency hiding).
        if (kc + 1 < KCHUNKS) load_chunk((kc + 1) * TILE_K);

        // Fragments per ISA 16-bit VGPR layouts:
        // A 16x32: lane half 0 holds K 0-7 & 16-23; half 1 holds K 8-15 & 24-31.
        // B 32x16: same K split as A (half 0: K 0-7 & 16-23; half 1: K 8-15 & 24-31).
        union Frag { uint4 q[2]; v16bf v; };
        Frag a[2], b[4];
#pragma unroll
        for (int mi = 0; mi < 2; ++mi) {
            const uint4* pa = (const uint4*)(&lds_a[cur][(wm + mi * 16 + l16) * TILE_K + lh * 8]);
            a[mi].q[0] = pa[0];           // K = lh*8 .. +7
            a[mi].q[1] = pa[2];           // K = 16+lh*8 .. +7
        }
#pragma unroll
        for (int ni = 0; ni < 4; ++ni) {
            const uint4* pb = (const uint4*)(&lds_b[cur][(wn + ni * 16 + l16) * TILE_K + lh * 8]);
            b[ni].q[0] = pb[0];           // K = lh*8 .. +7
            b[ni].q[1] = pb[2];           // K = 16+lh*8 .. +7
        }

#pragma unroll
        for (int mi = 0; mi < 2; ++mi)
#pragma unroll
            for (int ni = 0; ni < 4; ++ni)
                acc[mi][ni] = __builtin_amdgcn_wmma_f32_16x16x32_bf16(
                        false, a[mi].v, false, b[ni].v,
                        (short)0, acc[mi][ni], false, false);

        // Convert+store next chunk into the OTHER buffer (no read conflict),
        // then one barrier makes it visible for the next iteration.
        if (kc + 1 < KCHUNKS) store_chunk(cur ^ 1);
        __syncthreads();
    }

    // Epilogue: C/D layout is VGPR r -> row lh*8+r, col l16.
#pragma unroll
    for (int mi = 0; mi < 2; ++mi)
#pragma unroll
        for (int ni = 0; ni < 4; ++ni) {
            const int col = n0 + wn + ni * 16 + l16;
            const int rowb = m0 + wm + mi * 16 + lh * 8;
#pragma unroll
            for (int r = 0; r < 8; ++r)
                h1[(size_t)(rowb + r) * O + col] = acc[mi][ni][r];
        }
}

// ---------------------------------------------------------------------------
// LIF scan: one thread per (b, o) chain; syn/mem live in registers.
// Each wave touches one contiguous 128B line per timestep (coalesced).
// rst == spike (same predicate), so one compare feeds both uses.
// ---------------------------------------------------------------------------
__global__ __launch_bounds__(256) void snn_lif_scan(
        const float* __restrict__ h1,
        float* __restrict__ out) {
    const int idx = blockIdx.x * 256 + threadIdx.x;   // 0 .. B*O-1
    const int b = idx >> 9;                           // /512
    const int o = idx & (O - 1);
    const float* hp = h1  + (size_t)b * T * O + o;
    float*       op = out + (size_t)b * T * O + o;

    float syn = 0.0f, mem = 0.0f;
#pragma unroll 2
    for (int t = 0; t < T; ++t) {
        const float h    = hp[(size_t)t * O];
        const float mthr = mem - 1.0f;
        const float s    = (mthr > 0.0f) ? 1.0f : 0.0f;   // spike == rst
        op[(size_t)t * O] = s;
        const float nsyn = ALPHA * syn + h;
        mem = BETA * mem + syn - s;                        // uses OLD syn
        syn = nsyn;
    }
}

extern "C" void kernel_launch(void* const* d_in, const int* in_sizes, int n_in,
                              void* d_out, int out_size, void* d_ws, size_t ws_size,
                              hipStream_t stream) {
    const float* x = (const float*)d_in[0];   // [B, T, I] f32
    const float* w = (const float*)d_in[1];   // [I, O]    f32
    float* out = (float*)d_out;               // [B, T, O] f32
    float* h1  = (float*)d_ws;                // [M, O] f32 scratch (65.5 MB)

    dim3 gemm_grid(O / TILE_N, M / TILE_M);   // (4, 250)
    snn_gemm_bf16<<<gemm_grid, 256, 0, stream>>>(x, w, h1);

    const int chains = Bsz * O;               // 65536
    snn_lif_scan<<<chains / 256, 256, 0, stream>>>(h1, out);
}
